// GRU_83434034692133
// MI455X (gfx1250) — matrix operand order repaired and checked by reading, not compiled, not run on hardware
//
#include <hip/hip_runtime.h>
#include <hip/hip_bf16.h>

typedef __attribute__((ext_vector_type(16))) _Float16 v16h;
typedef __attribute__((ext_vector_type(8)))  _Float16 v8h;
typedef __attribute__((ext_vector_type(8)))  float    v8f;

#define LB_CONST 0.0f
#define UB_CONST 0.2f

// ---------------------------------------------------------------------------
// helpers
// ---------------------------------------------------------------------------
__device__ __forceinline__ float sigmoidf_(float x) {
    return 1.0f / (1.0f + __expf(-x));
}

// Load one lane's A-fragment for v_wmma_f32_16x16x32_f16.
// p must already point at  A_row_base + hi*16  (halves); kk added by caller.
// Lane(hi=0): halves[0..15]=K 0..15
// Lane(hi=1): halves[0..15]=K 16..31
__device__ __forceinline__ v16h load_a_frag(const _Float16* __restrict__ p) {
    v8h lo = *reinterpret_cast<const v8h*>(p);
    v8h hi = *reinterpret_cast<const v8h*>(p + 8);
    v16h out;
#pragma unroll
    for (int i = 0; i < 8; ++i) { out[i] = lo[i]; out[i + 8] = hi[i]; }
    return out;
}

__device__ __forceinline__ v8f wmma_f16(v16h a, v16h b, v8f c) {
    return __builtin_amdgcn_wmma_f32_16x16x32_f16(false, a, false, b,
                                                  (short)0, c, false, false);
}

// block-wide reductions (256 threads)
__device__ __forceinline__ float block_sum(float v, float* red) {
    int tid = threadIdx.x;
    red[tid] = v; __syncthreads();
#pragma unroll
    for (int s = 128; s > 0; s >>= 1) {
        if (tid < s) red[tid] += red[tid + s];
        __syncthreads();
    }
    float r = red[0]; __syncthreads();
    return r;
}
__device__ __forceinline__ float block_max(float v, float* red) {
    int tid = threadIdx.x;
    red[tid] = v; __syncthreads();
#pragma unroll
    for (int s = 128; s > 0; s >>= 1) {
        if (tid < s) red[tid] = fmaxf(red[tid], red[tid + s]);
        __syncthreads();
    }
    float r = red[0]; __syncthreads();
    return r;
}

// ---------------------------------------------------------------------------
// elementwise utility kernels
// ---------------------------------------------------------------------------
__global__ void f32_to_f16_kernel(const float* __restrict__ in,
                                  _Float16* __restrict__ out, long long n) {
    long long stride = (long long)gridDim.x * blockDim.x;
    for (long long i = (long long)blockIdx.x * blockDim.x + threadIdx.x;
         i < n; i += stride)
        out[i] = (_Float16)in[i];
}

__global__ void zero_f16_kernel(_Float16* __restrict__ p, long long n) {
    long long stride = (long long)gridDim.x * blockDim.x;
    for (long long i = (long long)blockIdx.x * blockDim.x + threadIdx.x;
         i < n; i += stride)
        p[i] = (_Float16)0.0f;
}

// ---------------------------------------------------------------------------
// WMMA GEMM:  out[M][N] = A[M][K] @ W[N][K]^T + bias[N]
// Each wave computes a 32(M) x 64(N) tile: 2 A-frags x 4 B-frags -> 8 WMMAs
// per 32-wide K-chunk with full register reuse of fragments.
// Requires M%32==0, N%64==0, K%32==0, and waves % (blockDim/32) == 0.
// ---------------------------------------------------------------------------
__global__ void gemm_wmma_kernel(const _Float16* __restrict__ A,
                                 const _Float16* __restrict__ W,
                                 const float* __restrict__ bias,
                                 void* __restrict__ out,
                                 int M, int N, int K, int out_is_f32) {
    int wave = blockIdx.x * (blockDim.x >> 5) + (threadIdx.x >> 5);
    int lane = threadIdx.x & 31;
    int tiles_n = N >> 6;                 // 64-wide N tiles
    int tm = wave / tiles_n;
    int tn = wave - tm * tiles_n;
    int m0 = tm << 5, n0 = tn << 6;
    int r  = lane & 15;
    int hi = lane >> 4;

    const _Float16* arow0 = A + (size_t)(m0 + r) * K + hi * 16;
    const _Float16* arow1 = arow0 + (size_t)16 * K;
    const _Float16* brow0 = W + (size_t)(n0 + r) * K + hi * 16;
    const _Float16* brow1 = brow0 + (size_t)16 * K;
    const _Float16* brow2 = brow0 + (size_t)32 * K;
    const _Float16* brow3 = brow0 + (size_t)48 * K;

    v8f acc[2][4];
#pragma unroll
    for (int mi = 0; mi < 2; ++mi)
#pragma unroll
        for (int ni = 0; ni < 4; ++ni) acc[mi][ni] = (v8f){};

#pragma unroll 2
    for (int kk = 0; kk < K; kk += 32) {
        __builtin_prefetch(arow0 + kk + 256, 0, 1);
        __builtin_prefetch(brow0 + kk + 256, 0, 1);
        v16h a0 = load_a_frag(arow0 + kk);
        v16h a1 = load_a_frag(arow1 + kk);
        v16h b0 = *reinterpret_cast<const v16h*>(brow0 + kk);
        v16h b1 = *reinterpret_cast<const v16h*>(brow1 + kk);
        v16h b2 = *reinterpret_cast<const v16h*>(brow2 + kk);
        v16h b3 = *reinterpret_cast<const v16h*>(brow3 + kk);
        acc[0][0] = wmma_f16(a0, b0, acc[0][0]);
        acc[0][1] = wmma_f16(a0, b1, acc[0][1]);
        acc[0][2] = wmma_f16(a0, b2, acc[0][2]);
        acc[0][3] = wmma_f16(a0, b3, acc[0][3]);
        acc[1][0] = wmma_f16(a1, b0, acc[1][0]);
        acc[1][1] = wmma_f16(a1, b1, acc[1][1]);
        acc[1][2] = wmma_f16(a1, b2, acc[1][2]);
        acc[1][3] = wmma_f16(a1, b3, acc[1][3]);
    }

#pragma unroll
    for (int ni = 0; ni < 4; ++ni) {
        int n = n0 + (ni << 4) + r;
        float bn = bias ? bias[n] : 0.0f;
#pragma unroll
        for (int mi = 0; mi < 2; ++mi) {
#pragma unroll
            for (int v = 0; v < 8; ++v) {
                int m = m0 + (mi << 4) + v + (hi << 3);
                float val = acc[mi][ni][v] + bn;
                if (out_is_f32)
                    ((float*)out)[(size_t)m * N + n] = val;
                else
                    ((_Float16*)out)[(size_t)m * N + n] = (_Float16)val;
            }
        }
    }
}

// ---------------------------------------------------------------------------
// Fused GRU step:  hp = hprev @ Whh^T (3 gates) + combine with precomputed
// x-projection -> h_new.  hprev [B][H] f16, xp [B][T][3H] f16 (b_ih folded),
// whh [3H][H] f16, bhh [3H] f32.
// Each wave computes a 32(M) x 16(N) tile of h_new: 2 A-frags x 3 gate
// B-frags -> 6 WMMAs per K-chunk; B fragments reused across both M-tiles.
// ---------------------------------------------------------------------------
__global__ void gru_step_kernel(const _Float16* __restrict__ hprev,
                                const _Float16* __restrict__ xp,
                                const _Float16* __restrict__ whh,
                                const float* __restrict__ bhh,
                                _Float16* __restrict__ hout,
                                _Float16* __restrict__ hseq,   // may be null
                                int t) {
    const int H = 1024, T = 128;
    int wave = blockIdx.x * (blockDim.x >> 5) + (threadIdx.x >> 5);
    int lane = threadIdx.x & 31;
    int tm = wave >> 6;          // tiles_n = H/16 = 64
    int tn = wave & 63;
    int m0 = tm << 5, n0 = tn << 4;
    int r  = lane & 15;
    int hi = lane >> 4;

    const _Float16* arow0 = hprev + (size_t)(m0 + r) * H + hi * 16;
    const _Float16* arow1 = arow0 + (size_t)16 * H;
    const _Float16* br = whh + (size_t)(n0 + r) * H + hi * 16;
    const _Float16* bz = br + (size_t)H * H;
    const _Float16* bn = br + (size_t)2 * H * H;

    v8f accr[2] = {(v8f){}, (v8f){}};
    v8f accz[2] = {(v8f){}, (v8f){}};
    v8f accn[2] = {(v8f){}, (v8f){}};

#pragma unroll 2
    for (int kk = 0; kk < H; kk += 32) {
        __builtin_prefetch(arow0 + kk + 256, 0, 1);
        v16h a0 = load_a_frag(arow0 + kk);
        v16h a1 = load_a_frag(arow1 + kk);
        v16h b0 = *reinterpret_cast<const v16h*>(br + kk);
        v16h b1 = *reinterpret_cast<const v16h*>(bz + kk);
        v16h b2 = *reinterpret_cast<const v16h*>(bn + kk);
        accr[0] = wmma_f16(a0, b0, accr[0]);
        accz[0] = wmma_f16(a0, b1, accz[0]);
        accn[0] = wmma_f16(a0, b2, accn[0]);
        accr[1] = wmma_f16(a1, b0, accr[1]);
        accz[1] = wmma_f16(a1, b1, accz[1]);
        accn[1] = wmma_f16(a1, b2, accn[1]);
    }

    int n = n0 + r;
    float b_r = bhh[n], b_z = bhh[n + H], b_n = bhh[n + 2 * H];
#pragma unroll
    for (int mi = 0; mi < 2; ++mi) {
#pragma unroll
        for (int v = 0; v < 8; ++v) {
            int m = m0 + (mi << 4) + v + (hi << 3);
            size_t xoff = ((size_t)m * T + t) * (size_t)(3 * H) + n;
            float xr = (float)xp[xoff];
            float xz = (float)xp[xoff + H];
            float xn = (float)xp[xoff + 2 * H];
            float rr = sigmoidf_(xr + accr[mi][v] + b_r);
            float zz = sigmoidf_(xz + accz[mi][v] + b_z);
            float nn = tanhf(xn + rr * (accn[mi][v] + b_n));
            float hp = (float)hprev[(size_t)m * H + n];
            float hv = (1.0f - zz) * nn + zz * hp;
            hout[(size_t)m * H + n] = (_Float16)hv;
            if (hseq)
                hseq[((size_t)m * T + t) * (size_t)H + n] = (_Float16)hv;
        }
    }
}

// ---------------------------------------------------------------------------
// SiLU + row softmax over 1024 columns. One 256-thread block per row.
// ---------------------------------------------------------------------------
__global__ void silu_softmax_kernel(const float* __restrict__ logits,
                                    float* __restrict__ probs) {
    __shared__ float red[256];
    const int C = 1024;
    int row = blockIdx.x, tid = threadIdx.x;
    float v[4];
#pragma unroll
    for (int i = 0; i < 4; ++i) {
        float l = logits[(size_t)row * C + tid + i * 256];
        v[i] = l * sigmoidf_(l);                       // SiLU
    }
    float m = fmaxf(fmaxf(v[0], v[1]), fmaxf(v[2], v[3]));
    m = block_max(m, red);
    float s = 0.0f;
#pragma unroll
    for (int i = 0; i < 4; ++i) { v[i] = __expf(v[i] - m); s += v[i]; }
    s = block_sum(s, red);
    float inv = 1.0f / s;
#pragma unroll
    for (int i = 0; i < 4; ++i)
        probs[(size_t)row * C + tid + i * 256] = v[i] * inv;
}

// ---------------------------------------------------------------------------
// Iterative rebalance (32 iterations), one 256-thread block per row of 1024.
// ---------------------------------------------------------------------------
__global__ void rebalance_kernel(const float* __restrict__ w_in,
                                 float* __restrict__ out, int n_iters) {
    __shared__ float red[256];
    const int C = 1024;
    int row = blockIdx.x, tid = threadIdx.x;
    float oldv[4], wc[4];
#pragma unroll
    for (int i = 0; i < 4; ++i) {
        float w = w_in[(size_t)row * C + tid + i * 256];
        oldv[i] = w;
        wc[i]   = fminf(fmaxf(w, LB_CONST), UB_CONST);
    }
    bool done = false;
    for (int it = 0; it < n_iters; ++it) {
        float diff = 0.0f, cnt = 0.0f, nom = 0.0f;
#pragma unroll
        for (int i = 0; i < 4; ++i) {
            diff += oldv[i] - wc[i];
            bool mask = (wc[i] != UB_CONST);
            cnt += mask ? 1.0f : 0.0f;
            nom += mask ? wc[i] : 0.0f;
        }
        float leftover = block_sum(diff, red);
        float num      = block_sum(cnt, red);
        float nomsum   = block_sum(nom, red);
        float denom = (nomsum == 0.0f) ? 1.0f : nomsum;

        float wcn[4], mx = -1e30f;
#pragma unroll
        for (int i = 0; i < 4; ++i) {
            bool mask = (wc[i] != UB_CONST);
            float gift = leftover * wc[i] / denom;
            wcn[i] = mask ? (wc[i] + gift) : wc[i];
            mx = fmaxf(mx, wcn[i]);
        }
        float gmax = block_max(mx, red);
        bool no_nom = (num == 0.0f);
        bool over   = (gmax > UB_CONST);
        bool new_done = done || no_nom || !over;
        if (!done && !no_nom) {
#pragma unroll
            for (int i = 0; i < 4; ++i) {
                oldv[i] = wcn[i];
                wc[i] = over ? fminf(fmaxf(wcn[i], LB_CONST), UB_CONST) : wcn[i];
            }
        }
        done = new_done;
    }
#pragma unroll
    for (int i = 0; i < 4; ++i)
        out[(size_t)row * C + tid + i * 256] = wc[i];
}

// ---------------------------------------------------------------------------
// kernel_launch
// ---------------------------------------------------------------------------
extern "C" void kernel_launch(void* const* d_in, const int* in_sizes, int n_in,
                              void* d_out, int out_size, void* d_ws, size_t ws_size,
                              hipStream_t stream) {
    (void)in_sizes; (void)n_in; (void)out_size; (void)ws_size;
    const float* x      = (const float*)d_in[0];
    const float* w_ih0  = (const float*)d_in[1];
    const float* w_hh0  = (const float*)d_in[2];
    const float* b_ih0  = (const float*)d_in[3];
    const float* b_hh0  = (const float*)d_in[4];
    const float* w_ih1  = (const float*)d_in[5];
    const float* w_hh1  = (const float*)d_in[6];
    const float* b_ih1  = (const float*)d_in[7];
    const float* b_hh1  = (const float*)d_in[8];
    const float* fc_w   = (const float*)d_in[9];
    const float* fc_b   = (const float*)d_in[10];

    const int B = 512, T = 128, H = 1024, S = 1024;
    const size_t M = (size_t)B * T;                     // 65536 rows

    // workspace carve-out (256B aligned)
    char* ws = (char*)d_ws;
    size_t off = 0;
    auto take = [&](size_t bytes) -> char* {
        char* p = ws + off;
        off = (off + bytes + 255) & ~(size_t)255;
        return p;
    };
    _Float16* X16  = (_Float16*)take(M * S * 2);                 // x in f16
    _Float16* WIH0 = (_Float16*)take((size_t)3 * H * S * 2);
    _Float16* WHH0 = (_Float16*)take((size_t)3 * H * H * 2);
    _Float16* WIH1 = (_Float16*)take((size_t)3 * H * H * 2);
    _Float16* WHH1 = (_Float16*)take((size_t)3 * H * H * 2);
    _Float16* FCW  = (_Float16*)take((size_t)S * H * 2);
    _Float16* XP   = (_Float16*)take(M * (size_t)3 * H * 2);     // x-proj (reused)
    _Float16* HSEQ = (_Float16*)take(M * (size_t)H * 2);         // layer0 outputs
    _Float16* HB0  = (_Float16*)take((size_t)B * H * 2);         // h ping
    _Float16* HB1  = (_Float16*)take((size_t)B * H * 2);         // h pong
    float*    LOGITS = (float*)take((size_t)B * S * 4);
    float*    PROBS  = (float*)take((size_t)B * S * 4);

    // 1) f32 -> f16 converts
    auto conv = [&](const float* src, _Float16* dst, long long n) {
        f32_to_f16_kernel<<<2048, 256, 0, stream>>>(src, dst, n);
    };
    conv(x,     X16,  (long long)M * S);
    conv(w_ih0, WIH0, (long long)3 * H * S);
    conv(w_hh0, WHH0, (long long)3 * H * H);
    conv(w_ih1, WIH1, (long long)3 * H * H);
    conv(w_hh1, WHH1, (long long)3 * H * H);
    conv(fc_w,  FCW,  (long long)S * H);

    const int wavesPerBlock = 8;

    // 2) layer-0 x-projection: XP = X16 @ WIH0^T + b_ih0   (65536 x 3072 x 1024)
    {
        size_t waves = (M / 32) * ((size_t)(3 * H) / 64);        // 98304
        gemm_wmma_kernel<<<(unsigned)(waves / wavesPerBlock), 256, 0, stream>>>(
            X16, WIH0, b_ih0, XP, (int)M, 3 * H, S, 0);
    }

    // 3) layer-0 recurrence
    zero_f16_kernel<<<512, 256, 0, stream>>>(HB0, (long long)B * H);
    for (int t = 0; t < T; ++t) {
        _Float16* hprev = (t & 1) ? HB1 : HB0;
        _Float16* hout  = (t & 1) ? HB0 : HB1;
        gru_step_kernel<<<(B / 32) * (H / 16) / wavesPerBlock, 256, 0, stream>>>(
            hprev, XP, WHH0, b_hh0, hout, HSEQ, t);
    }

    // 4) layer-1 x-projection: XP = HSEQ @ WIH1^T + b_ih1
    {
        size_t waves = (M / 32) * ((size_t)(3 * H) / 64);
        gemm_wmma_kernel<<<(unsigned)(waves / wavesPerBlock), 256, 0, stream>>>(
            HSEQ, WIH1, b_ih1, XP, (int)M, 3 * H, H, 0);
    }

    // 5) layer-1 recurrence (sequence not needed)
    zero_f16_kernel<<<512, 256, 0, stream>>>(HB0, (long long)B * H);
    for (int t = 0; t < T; ++t) {
        _Float16* hprev = (t & 1) ? HB1 : HB0;
        _Float16* hout  = (t & 1) ? HB0 : HB1;
        gru_step_kernel<<<(B / 32) * (H / 16) / wavesPerBlock, 256, 0, stream>>>(
            hprev, XP, WHH1, b_hh1, hout, nullptr, t);
    }
    // after t = 127, final hidden state lives in HB0 (128 is even)

    // 6) FC head: LOGITS = HB0 @ FCW^T + fc_b  (512 x 1024 x 1024, f32 out)
    {
        size_t waves = ((size_t)B / 32) * ((size_t)S / 64);      // 256
        gemm_wmma_kernel<<<(unsigned)(waves / wavesPerBlock), 256, 0, stream>>>(
            HB0, FCW, fc_b, LOGITS, B, S, H, 1);
    }

    // 7) SiLU + softmax per row
    silu_softmax_kernel<<<B, 256, 0, stream>>>(LOGITS, PROBS);

    // 8) 32-iteration rebalance -> d_out
    rebalance_kernel<<<B, 256, 0, stream>>>(PROBS, (float*)d_out, 32);
}
